// ResCapsNet_87153476370888
// MI455X (gfx1250) — matrix operand order repaired and checked by reading, not compiled, not run on hardware
//
#include <hip/hip_runtime.h>
#include <hip/hip_bf16.h>

// ---------------------------------------------------------------------------
// ResCapsNet forward on gfx1250 (MI455X, wave32, WMMA).
// Heavy GEMMs: v_wmma_f32_16x16x32_bf16 (bf16 in, fp32 accumulate).
// LDS weight staging: global_load_async_to_lds_b128 + s_wait_asynccnt.
// ---------------------------------------------------------------------------

typedef __attribute__((ext_vector_type(16))) __bf16       v16bf;
typedef __attribute__((ext_vector_type(8)))  float        v8f;
typedef __attribute__((ext_vector_type(4)))  unsigned int u32x4;

union Frag {
    v16bf  v;
    u32x4  q[2];
    __bf16 h[16];
};

static __device__ inline v8f zero8() {
    v8f z;
#pragma unroll
    for (int i = 0; i < 8; ++i) z[i] = 0.0f;
    return z;
}
static __device__ inline u32x4 zero4() {
    u32x4 z;
#pragma unroll
    for (int i = 0; i < 4; ++i) z[i] = 0u;
    return z;
}

static __device__ inline v8f wmma_bf16(const Frag& a, const Frag& b, v8f c) {
    // (neg_a, A, neg_b, B, c_mod, C, reuse_a, reuse_b)
    return __builtin_amdgcn_wmma_f32_16x16x32_bf16(false, a.v, false, b.v,
                                                   (short)0, c, false, false);
}

// --- CDNA5 async global->LDS copy (16B per lane), tracked by ASYNCcnt -------
// LDS flat addresses: aperture in addr[63:32], so low 32 bits are the LDS
// byte offset (ISA 10.2 aperture rules).
static __device__ inline void async_copy_b128(void* lds_dst, const void* gsrc) {
    unsigned ldsoff = (unsigned)(size_t)lds_dst;
    asm volatile("global_load_async_to_lds_b128 %0, %1, off"
                 :: "v"(ldsoff), "v"(gsrc) : "memory");
}
static __device__ inline void wait_async0() {
    asm volatile("s_wait_asynccnt 0x0" ::: "memory");
}

// ------------------------------ sizes --------------------------------------
#define BATCH     128
#define HW        80
#define NPIX      (BATCH * HW * HW)          // 819200 pixels
#define NPRIM     1296
#define NCLS      10
#define DDIM      16
#define NPPIX     (BATCH * 81)               // 10368 primary-caps pixels

// ---------------------------------------------------------------------------
// conv1: 1->32, 3x3, pad 1, + bias, NHWC fp32 output.
// ---------------------------------------------------------------------------
__global__ void k_conv1(const float* __restrict__ x, const float* __restrict__ w,
                        const float* __restrict__ bias, float* __restrict__ out) {
    __shared__ float lw[288];
    __shared__ float lb[32];
    int tid = threadIdx.x;
    if (tid < 288) lw[tid] = w[tid];
    if (tid < 32)  lb[tid] = bias[tid];
    __syncthreads();
    int pix = blockIdx.x * 256 + tid;         // NPIX == 3200*256 exactly
    int b = pix / 6400, rem = pix % 6400;
    int y = rem / 80,   xx  = rem % 80;
    const float* xb = x + (size_t)b * 6400;
    float in9[9];
#pragma unroll
    for (int t = 0; t < 9; ++t) {
        int yy = y + t / 3 - 1, xc = xx + t % 3 - 1;
        bool ok = ((unsigned)yy < 80u) && ((unsigned)xc < 80u);
        in9[t] = ok ? xb[yy * 80 + xc] : 0.0f;
    }
    float* op = out + (size_t)pix * 32;
#pragma unroll
    for (int c = 0; c < 32; ++c) {
        float acc = lb[c];
#pragma unroll
        for (int t = 0; t < 9; ++t) acc += in9[t] * lw[c * 9 + t];
        op[c] = acc;
    }
}

// ---------------------------------------------------------------------------
// Batch-norm statistics over (B,H,W) per channel: sum / sumsq via atomics.
// ---------------------------------------------------------------------------
__global__ void k_bn_stats(const float* __restrict__ xin, float* __restrict__ stats,
                           int npix) {
    __shared__ float rs[256], rq[256];
    int tid = threadIdx.x;
    int ch = tid & 31, grp = tid >> 5;        // 8 pixel-groups x 32 channels
    float s = 0.0f, q = 0.0f;
    for (int pix = blockIdx.x * 8 + grp; pix < npix; pix += gridDim.x * 8) {
        float v = xin[(size_t)pix * 32 + ch];
        s += v; q += v * v;
    }
    rs[tid] = s; rq[tid] = q;
    __syncthreads();
    if (grp == 0) {
        float ts = 0.0f, tq = 0.0f;
#pragma unroll
        for (int g = 0; g < 8; ++g) { ts += rs[g * 32 + ch]; tq += rq[g * 32 + ch]; }
        atomicAdd(&stats[ch], ts);
        atomicAdd(&stats[32 + ch], tq);
    }
}

__global__ void k_bn_finalize(const float* __restrict__ stats,
                              const float* __restrict__ g, const float* __restrict__ b,
                              float* __restrict__ ss, float inv_n) {
    int c = threadIdx.x;
    if (c < 32) {
        float mean = stats[c] * inv_n;
        float var  = stats[32 + c] * inv_n - mean * mean;   // biased, matches jnp.var
        float sc   = g[c] * rsqrtf(var + 1e-5f);
        ss[c]      = sc;
        ss[32 + c] = b[c] - mean * sc;
    }
}

// BN apply (+ optional residual) + ReLU; writes bf16 (GEMM input) and optional fp32.
__global__ void k_bn_apply(const float* __restrict__ xin, const float* __restrict__ idt,
                           const float* __restrict__ ss, float* __restrict__ outf,
                           __bf16* __restrict__ outb, int n) {
    int t = blockIdx.x * 256 + threadIdx.x;
    if (t >= n) return;
    int ch = t & 31;
    float v = xin[t] * ss[ch] + ss[32 + ch];
    if (idt) v += idt[t];
    v = fmaxf(v, 0.0f);
    if (outf) outf[t] = v;
    outb[t] = (__bf16)v;
}

// ---------------------------------------------------------------------------
// Weight converters / repackers (fp32 -> bf16, GEMM-friendly layouts).
// ---------------------------------------------------------------------------
__global__ void k_cvt_bf16(const float* __restrict__ in, __bf16* __restrict__ out, int n) {
    int t = blockIdx.x * 256 + threadIdx.x;
    if (t < n) out[t] = (__bf16)in[t];
}

// rb_c2_w [blk][o][i][ky][kx] -> [blk][tap][o][i]
__global__ void k_repack_rb2(const float* __restrict__ in, __bf16* __restrict__ out) {
    int t = blockIdx.x * 256 + threadIdx.x;
    if (t >= 3 * 32 * 32 * 9) return;
    int blk = t / 9216, r = t % 9216;
    int o = r / 288, r2 = r % 288;
    int i = r2 / 9, tap = r2 % 9;
    out[(((size_t)blk * 9 + tap) * 32 + o) * 32 + i] = (__bf16)in[t];
}

// pcaps_w [o128][i32][ky9][kx9] -> [tap81][o128][i32]
__global__ void k_repack_pc(const float* __restrict__ in, __bf16* __restrict__ out) {
    int t = blockIdx.x * 256 + threadIdx.x;
    if (t >= 128 * 32 * 81) return;
    int o = t / 2592, r = t % 2592;
    int i = r / 81, tap = r % 81;
    out[((size_t)tap * 128 + o) * 32 + i] = (__bf16)in[t];
}

// ---------------------------------------------------------------------------
// 1x1 conv as GEMM: [NPIX,32] x [32,32] -> [NPIX,32].  W layout [out][in].
// Block = 8 waves; wave computes a 16(M) x 32(N) tile in two WMMAs (K=32).
// ---------------------------------------------------------------------------
__global__ void k_gemm1x1(const __bf16* __restrict__ A, const __bf16* __restrict__ W,
                          float* __restrict__ out) {
    __shared__ __attribute__((aligned(16))) __bf16 lw[1024];
    int tid = threadIdx.x;
    if (tid < 128)                                   // 128 lanes x 16B = 2KB
        async_copy_b128(&lw[tid * 8], W + tid * 8);
    wait_async0();
    __syncthreads();
    int wav = tid >> 5, lane = tid & 31;
    int half = lane >> 4, lr = lane & 15;
    int mbase = blockIdx.x * 128 + wav * 16;
    int row = mbase + lr;
    Frag a;
    a.q[0] = *(const u32x4*)(A + (size_t)row * 32 + half * 16);       // K 0-7 / 16-23
    a.q[1] = *(const u32x4*)(A + (size_t)row * 32 + half * 16 + 8);   // K 8-15 / 24-31
    Frag b0, b1;
    const u32x4* bp0 = (const u32x4*)&lw[(0 + lr) * 32 + half * 16];
    b0.q[0] = bp0[0]; b0.q[1] = bp0[1];
    const u32x4* bp1 = (const u32x4*)&lw[(16 + lr) * 32 + half * 16];
    b1.q[0] = bp1[0]; b1.q[1] = bp1[1];
    v8f acc0 = zero8(), acc1 = zero8();
    acc0 = wmma_bf16(a, b0, acc0);
    acc1 = wmma_bf16(a, b1, acc1);
#pragma unroll
    for (int i = 0; i < 8; ++i) {
        int r = mbase + i + half * 8;
        out[(size_t)r * 32 + lr]      = acc0[i];
        out[(size_t)r * 32 + 16 + lr] = acc1[i];
    }
}

// ---------------------------------------------------------------------------
// 3x3 conv (pad 1) as implicit GEMM over 9 taps.  W layout [tap][out][in] bf16.
// Weights staged via async global->LDS (576 x 16B chunks = 4608 elems).
// ---------------------------------------------------------------------------
__global__ void k_conv3x3(const __bf16* __restrict__ A, const __bf16* __restrict__ W,
                          float* __restrict__ out) {
    __shared__ __attribute__((aligned(16))) __bf16 lw[9 * 1024];
    int tid = threadIdx.x;
    for (int i = tid; i < 576; i += 256)
        async_copy_b128(&lw[i * 8], W + (size_t)i * 8);
    wait_async0();
    __syncthreads();
    int wav = tid >> 5, lane = tid & 31;
    int half = lane >> 4, lr = lane & 15;
    int mbase = blockIdx.x * 128 + wav * 16;
    int pix = mbase + lr;
    int b = pix / 6400, rem = pix % 6400;
    int y = rem / 80, x = rem % 80;
    v8f acc0 = zero8(), acc1 = zero8();
#pragma unroll
    for (int t = 0; t < 9; ++t) {
        int yy = y + t / 3 - 1, xx = x + t % 3 - 1;
        Frag a;
        if (((unsigned)yy < 80u) && ((unsigned)xx < 80u)) {
            const __bf16* ap = A + (((size_t)b * 80 + yy) * 80 + xx) * 32;
            a.q[0] = *(const u32x4*)(ap + half * 16);
            a.q[1] = *(const u32x4*)(ap + half * 16 + 8);
        } else {
            a.q[0] = zero4(); a.q[1] = zero4();
        }
        Frag b0, b1;
        const u32x4* bp0 = (const u32x4*)&lw[t * 1024 + lr * 32 + half * 16];
        b0.q[0] = bp0[0]; b0.q[1] = bp0[1];
        const u32x4* bp1 = (const u32x4*)&lw[t * 1024 + (16 + lr) * 32 + half * 16];
        b1.q[0] = bp1[0]; b1.q[1] = bp1[1];
        acc0 = wmma_bf16(a, b0, acc0);
        acc1 = wmma_bf16(a, b1, acc1);
    }
#pragma unroll
    for (int i = 0; i < 8; ++i) {
        int r = mbase + i + half * 8;
        out[(size_t)r * 32 + lr]      = acc0[i];
        out[(size_t)r * 32 + 16 + lr] = acc1[i];
    }
}

// ---------------------------------------------------------------------------
// Primary-caps conv: 9x9 stride 8, 32 -> 128 channels, implicit GEMM over 81
// taps.  Block = 8 waves = 8 N-tiles of 16; one M-tile (16 pixels) per block.
// W layout [tap][out128][in32] bf16; per-tap 8KB tile double-buffered in LDS
// via async global->LDS copies overlapped with the previous tap's WMMA.
// Per-wave s_wait_asynccnt only covers that wave's copies; the barrier after
// the wait is what makes the full tile visible to all 8 waves.
// ---------------------------------------------------------------------------
__global__ void k_pcaps(const __bf16* __restrict__ A, const __bf16* __restrict__ W,
                        float* __restrict__ out) {
    __shared__ __attribute__((aligned(16))) __bf16 lw[2][4096];
    int tid = threadIdx.x;
    int wav = tid >> 5, lane = tid & 31;
    int half = lane >> 4, lr = lane & 15;
    int n0 = wav * 16;
    int pix = blockIdx.x * 16 + lr;
    int b = pix / 81, rem = pix % 81;
    int oy = rem / 9, ox = rem % 9;

    // prologue: stage tap 0 into buffer 0 (each thread copies 2 x 16B)
    async_copy_b128(&lw[0][tid * 8],        W + tid * 8);
    async_copy_b128(&lw[0][2048 + tid * 8], W + 2048 + tid * 8);

    v8f acc = zero8();
    for (int t = 0; t < 81; ++t) {
        int cur = t & 1;
        wait_async0();              // copies issued last iteration (tap t) done
        __syncthreads();            // ...for every wave; prev buffer free
        if (t < 80) {               // overlap: stage tap t+1 while computing t
            const __bf16* gw = W + (size_t)(t + 1) * 4096;
            async_copy_b128(&lw[cur ^ 1][tid * 8],        gw + tid * 8);
            async_copy_b128(&lw[cur ^ 1][2048 + tid * 8], gw + 2048 + tid * 8);
        }
        int ky = t / 9, kx = t % 9;
        const __bf16* ap = A + (((size_t)b * 80 + oy * 8 + ky) * 80 + ox * 8 + kx) * 32;
        Frag a;
        a.q[0] = *(const u32x4*)(ap + half * 16);
        a.q[1] = *(const u32x4*)(ap + half * 16 + 8);
        Frag bb;
        const u32x4* bp = (const u32x4*)&lw[cur][(n0 + lr) * 32 + half * 16];
        bb.q[0] = bp[0]; bb.q[1] = bp[1];
        acc = wmma_bf16(a, bb, acc);
    }
#pragma unroll
    for (int i = 0; i < 8; ++i) {
        int r = blockIdx.x * 16 + i + half * 8;
        out[(size_t)r * 128 + n0 + lr] = acc[i];
    }
}

// pcaps output [B*81,128] (+bias) -> u_bf [B][1296][8] bf16
__global__ void k_uprep(const float* __restrict__ p, const float* __restrict__ bias,
                        __bf16* __restrict__ ub) {
    int t = blockIdx.x * 256 + threadIdx.x;
    if (t >= BATCH * NPRIM * 8) return;
    int k = t & 7;
    int prim = (t >> 3) % NPRIM;
    int b = t / (NPRIM * 8);
    int cap = prim / 81, sp = prim % 81;
    int oc = cap * 8 + k;
    float v = p[((size_t)b * 81 + sp) * 128 + oc] + bias[oc];
    ub[t] = (__bf16)v;
}

// ---------------------------------------------------------------------------
// u_hat[b,p,c,d] = W[p,cd,k] * u[b,p,k].  Per p: M=128(b), N=160(cd), K=8(pad32).
// Wc layout is exactly W_caps flat ([p][cd][k]) converted to bf16.
// ---------------------------------------------------------------------------
__global__ void k_uhat(const __bf16* __restrict__ ub, const __bf16* __restrict__ Wc,
                       float* __restrict__ uhat) {
    int tid = threadIdx.x;
    int wav = tid >> 5, lane = tid & 31;
    int half = lane >> 4, lr = lane & 15;
    int p = blockIdx.x / 10;
    int tile = (blockIdx.x % 10) * 8 + wav;   // 0..79 = 8 Mtiles x 10 Ntiles
    int mt = tile / 10, nt = tile % 10;
    int mbase = mt * 16, n0 = nt * 16;
    Frag a, b;
    if (half == 0) {                          // lanes 0-15 carry K=0..7; rest zero
        a.q[0] = *(const u32x4*)(ub + ((size_t)(mbase + lr) * NPRIM + p) * 8);
        a.q[1] = zero4();
        b.q[0] = *(const u32x4*)(Wc + ((size_t)p * 160 + n0 + lr) * 8);
        b.q[1] = zero4();
    } else {
        a.q[0] = zero4(); a.q[1] = zero4();
        b.q[0] = zero4(); b.q[1] = zero4();
    }
    v8f acc = zero8();
    acc = wmma_bf16(a, b, acc);
#pragma unroll
    for (int i = 0; i < 8; ++i) {
        int r = mbase + i + half * 8;
        uhat[((size_t)r * NPRIM + p) * 160 + n0 + lr] = acc[i];
    }
}

// ---------------------------------------------------------------------------
// Routing: s[b,c,d] = sum_p softmax_c(b_log[b,p,:]) * u_hat[b,p,c,d]
// ---------------------------------------------------------------------------
__global__ void k_route_s(const float* __restrict__ blog, const float* __restrict__ uhat,
                          float* __restrict__ s) {
    int tid = threadIdx.x;                    // 160 threads: (c,d)
    int b = blockIdx.x >> 3;
    int chunk = blockIdx.x & 7;
    int c = tid / 16;
    float acc = 0.0f;
    int p0 = chunk * 162;
    for (int p = p0; p < p0 + 162; ++p) {
        const float* bl = blog + ((size_t)b * NPRIM + p) * NCLS;
        float m = bl[0];
#pragma unroll
        for (int j = 1; j < NCLS; ++j) m = fmaxf(m, bl[j]);
        float den = 0.0f;
#pragma unroll
        for (int j = 0; j < NCLS; ++j) den += __expf(bl[j] - m);
        float w = __expf(bl[c] - m) / den;
        acc += w * uhat[((size_t)b * NPRIM + p) * 160 + tid];
    }
    atomicAdd(&s[(size_t)b * 160 + tid], acc);
}

// squash(s) -> v ; final iteration also writes v and lengths into d_out
__global__ void k_squash(const float* __restrict__ s, float* __restrict__ vbuf,
                         float* __restrict__ outv, float* __restrict__ outlen) {
    int tid = threadIdx.x;
    int idx = blockIdx.x * 16 + (tid >> 4);   // (b,c) pair; 1280 total
    int d = tid & 15;
    float sv = s[(size_t)idx * 16 + d];
    float n2 = sv * sv;
#pragma unroll
    for (int off = 8; off >= 1; off >>= 1) n2 += __shfl_xor(n2, off, 16);
    float v = sv * (n2 / (1.0f + n2)) * rsqrtf(n2 + 1e-9f);
    vbuf[(size_t)idx * 16 + d] = v;
    if (outv)  outv[(size_t)idx * 16 + d] = v;
    if (outlen && d == 0) outlen[idx] = sqrtf(n2 + 1e-9f);
}

// b_log[b,p,c] += dot(u_hat[b,p,c,:], v[b,c,:])
__global__ void k_route_b(const float* __restrict__ uhat, const float* __restrict__ vbuf,
                          float* __restrict__ blog) {
    int t = blockIdx.x * 256 + threadIdx.x;
    if (t >= BATCH * NPRIM * NCLS) return;
    int c = t % NCLS;
    int p = (t / NCLS) % NPRIM;
    int b = t / (NPRIM * NCLS);
    const float* u = uhat + (((size_t)b * NPRIM + p) * NCLS + c) * 16;
    const float* v = vbuf + ((size_t)b * NCLS + c) * 16;
    float acc = 0.0f;
#pragma unroll
    for (int k = 0; k < 16; ++k) acc += u[k] * v[k];
    blog[t] += acc;
}

// ---------------------------------------------------------------------------
// Workspace layout (bytes, 256-aligned)
// ---------------------------------------------------------------------------
#define OFF_HF    ((size_t)0)                         // h fp32 NHWC     104857600
#define OFF_HB    (OFF_HF   + (size_t)104857600)      // h bf16           52428800
#define OFF_TF    (OFF_HB   + (size_t)52428800)       // tmp fp32        104857600
#define OFF_TB    (OFF_TF   + (size_t)104857600)      // tmp bf16         52428800
#define OFF_UHAT  (OFF_TB   + (size_t)52428800)       // u_hat fp32      106168320
#define OFF_UBF   (OFF_UHAT + (size_t)106168320)      // u bf16            2654208
#define OFF_BLOG  (OFF_UBF  + (size_t)2654208)        // routing logits    6635520
#define OFF_S     (OFF_BLOG + (size_t)6635520)        // s [B,10,16]         81920
#define OFF_V     (OFF_S    + (size_t)81920)          // v [B,10,16]         81920
#define OFF_STATS (OFF_V    + (size_t)81920)          // bn sum/sumsq          256
#define OFF_SS    (OFF_STATS+ (size_t)256)            // bn scale/shift        256
#define OFF_W1    (OFF_SS   + (size_t)256)            // rb 1x1 w bf16        6144
#define OFF_W2    (OFF_W1   + (size_t)6144)           // rb 3x3 w bf16       55296
#define OFF_WP    (OFF_W2   + (size_t)55296)          // pcaps w bf16       663552
#define OFF_WC    (OFF_WP   + (size_t)663552)         // W_caps bf16       3317760

extern "C" void kernel_launch(void* const* d_in, const int* in_sizes, int n_in,
                              void* d_out, int out_size, void* d_ws, size_t ws_size,
                              hipStream_t stream) {
    const float* x       = (const float*)d_in[0];
    const float* conv1_w = (const float*)d_in[1];
    const float* conv1_b = (const float*)d_in[2];
    const float* bn1_g   = (const float*)d_in[3];
    const float* bn1_b   = (const float*)d_in[4];
    const float* rb1_w   = (const float*)d_in[5];
    const float* rbg1    = (const float*)d_in[6];
    const float* rbb1    = (const float*)d_in[7];
    const float* rb2_w   = (const float*)d_in[8];
    const float* rbg2    = (const float*)d_in[9];
    const float* rbb2    = (const float*)d_in[10];
    const float* pc_w    = (const float*)d_in[11];
    const float* pc_b    = (const float*)d_in[12];
    const float* W_caps  = (const float*)d_in[13];

    char* ws = (char*)d_ws;
    float*  hf    = (float*)(ws + OFF_HF);
    __bf16* hb    = (__bf16*)(ws + OFF_HB);
    float*  tf    = (float*)(ws + OFF_TF);
    __bf16* tb    = (__bf16*)(ws + OFF_TB);
    float*  uhat  = (float*)(ws + OFF_UHAT);
    __bf16* ubf   = (__bf16*)(ws + OFF_UBF);
    float*  blog  = (float*)(ws + OFF_BLOG);
    float*  sbuf  = (float*)(ws + OFF_S);
    float*  vbuf  = (float*)(ws + OFF_V);
    float*  stats = (float*)(ws + OFF_STATS);
    float*  ssbuf = (float*)(ws + OFF_SS);
    __bf16* w1b   = (__bf16*)(ws + OFF_W1);
    __bf16* w2b   = (__bf16*)(ws + OFF_W2);
    __bf16* wpb   = (__bf16*)(ws + OFF_WP);
    __bf16* wcb   = (__bf16*)(ws + OFF_WC);

    const int   NELEM  = NPIX * 32;                 // 26,214,400
    const float INV_N  = 1.0f / (float)NPIX;

    // --- weight conversion / repacking (independent of activations) ---
    k_cvt_bf16   <<<12,   256, 0, stream>>>(rb1_w, w1b, 3 * 32 * 32);
    k_repack_rb2 <<<108,  256, 0, stream>>>(rb2_w, w2b);
    k_repack_pc  <<<1296, 256, 0, stream>>>(pc_w, wpb);
    k_cvt_bf16   <<<6480, 256, 0, stream>>>(W_caps, wcb, NPRIM * 160 * 8);

    // --- conv1 + BN + ReLU ---
    k_conv1<<<3200, 256, 0, stream>>>(x, conv1_w, conv1_b, tf);
    hipMemsetAsync(stats, 0, 256, stream);
    k_bn_stats   <<<512, 256, 0, stream>>>(tf, stats, NPIX);
    k_bn_finalize<<<1,   32,  0, stream>>>(stats, bn1_g, bn1_b, ssbuf, INV_N);
    k_bn_apply   <<<(NELEM + 255) / 256, 256, 0, stream>>>(tf, nullptr, ssbuf, hf, hb, NELEM);

    // --- residual blocks ---
    for (int blk = 0; blk < 3; ++blk) {
        // 1x1 conv -> BN -> ReLU (bf16 only)
        k_gemm1x1<<<6400, 256, 0, stream>>>(hb, w1b + (size_t)blk * 1024, tf);
        hipMemsetAsync(stats, 0, 256, stream);
        k_bn_stats   <<<512, 256, 0, stream>>>(tf, stats, NPIX);
        k_bn_finalize<<<1,   32,  0, stream>>>(stats, rbg1 + blk * 32, rbb1 + blk * 32,
                                               ssbuf, INV_N);
        k_bn_apply   <<<(NELEM + 255) / 256, 256, 0, stream>>>(tf, nullptr, ssbuf,
                                                               nullptr, tb, NELEM);
        // 3x3 conv -> BN -> +identity -> ReLU (fp32 + bf16)
        k_conv3x3<<<6400, 256, 0, stream>>>(tb, w2b + (size_t)blk * 9216, tf);
        hipMemsetAsync(stats, 0, 256, stream);
        k_bn_stats   <<<512, 256, 0, stream>>>(tf, stats, NPIX);
        k_bn_finalize<<<1,   32,  0, stream>>>(stats, rbg2 + blk * 32, rbb2 + blk * 32,
                                               ssbuf, INV_N);
        k_bn_apply   <<<(NELEM + 255) / 256, 256, 0, stream>>>(tf, hf, ssbuf, hf, hb, NELEM);
    }

    // --- primary capsules ---
    k_pcaps<<<NPPIX / 16, 256, 0, stream>>>(hb, wpb, tf);      // [10368,128] fp32
    k_uprep<<<(BATCH * NPRIM * 8 + 255) / 256, 256, 0, stream>>>(tf, pc_b, ubf);

    // --- u_hat GEMM ---
    k_uhat<<<NPRIM * 10, 256, 0, stream>>>(ubf, wcb, uhat);

    // --- dynamic routing (3 iterations) ---
    hipMemsetAsync(blog, 0, (size_t)BATCH * NPRIM * NCLS * 4, stream);
    float* outv   = (float*)d_out;                 // [B,10,16]
    float* outlen = (float*)d_out + BATCH * NCLS * DDIM;
    for (int r = 0; r < 3; ++r) {
        hipMemsetAsync(sbuf, 0, (size_t)BATCH * 160 * 4, stream);
        k_route_s<<<BATCH * 8, 160, 0, stream>>>(blog, uhat, sbuf);
        bool fin = (r == 2);
        k_squash<<<80, 256, 0, stream>>>(sbuf, vbuf,
                                         fin ? outv : nullptr,
                                         fin ? outlen : nullptr);
        if (!fin)
            k_route_b<<<(BATCH * NPRIM * NCLS + 255) / 256, 256, 0, stream>>>(uhat, vbuf, blog);
    }
}
